// AttentionMoE_40707700032217
// MI455X (gfx1250) — matrix operand order repaired and checked by reading, not compiled, not run on hardware
//
#include <hip/hip_runtime.h>
#include <math.h>
#include <stdint.h>

#define LQ 4096
#define DM 1024
#define NH 16
#define HDIM 64
#define NE 8
#define TOPK 2
#define HID 4096
// worst-case 16-aligned per-expert capacity: 8192 + 8*15 = 8312 -> 8448 (528 tiles)
#define NSLOT 8448
#define NTILES (NSLOT / 16)
#define ATT_SCALE 0.125f  // 1/sqrt(64)

typedef __attribute__((ext_vector_type(16))) __bf16 v16bf;
typedef __attribute__((ext_vector_type(8)))  float  v8f;

union BFrag { v16bf v; unsigned short u[16]; };

__device__ __forceinline__ unsigned short f2bf(float f) {
  union { float f; unsigned u; } c; c.f = f;
  unsigned u = c.u;
  u += 0x7FFFu + ((u >> 16) & 1u);   // round-to-nearest-even
  return (unsigned short)(u >> 16);
}
__device__ __forceinline__ float bf2f(unsigned short b) {
  union { unsigned u; float f; } c; c.u = ((unsigned)b) << 16;
  return c.f;
}
__device__ __forceinline__ v8f wmma_bf16(const BFrag& a, const BFrag& b, v8f c) {
  // D(f32 16x16) = A(bf16 16x32) * B(bf16 32x16) + C
  return __builtin_amdgcn_wmma_f32_16x16x32_bf16(false, a.v, false, b.v,
                                                 (short)0, c, false, false);
}

// Async global->LDS copy of a 32x64 f32 tile (8 KB). Rows strided by `stride`
// floats. Each of 16 issues moves 16 B/lane (512 B/wave) via
// GLOBAL_LOAD_ASYNC_TO_LDS_B128 (ASYNCcnt). lds_base = wave-relative LDS byte
// offset (low 32 bits of the generic shared pointer, per the aperture map).
__device__ __forceinline__ void async_tile_32x64(const float* __restrict__ g0,
                                                 int stride, unsigned lds_base,
                                                 int lane) {
  const int rsub = lane >> 4;        // 0/1: row within each 2-row slab
  const int c4   = (lane & 15) * 4;  // float column (16B granules)
  // WAR guard: prior in-order DS reads of this LDS region must retire before
  // the async unit can overwrite it.
  asm volatile("s_wait_dscnt 0" ::: "memory");
  for (int i = 0; i < 16; ++i) {
    const int row = i * 2 + rsub;
    const float* g = g0 + (size_t)row * stride + c4;
    const unsigned l = lds_base + (unsigned)(row * 64 + c4) * 4u;
    asm volatile("global_load_async_to_lds_b128 %0, %1, off"
                 :: "v"(l), "v"(g) : "memory");
  }
}
__device__ __forceinline__ void wait_async0() {
  asm volatile("s_wait_asynccnt 0" ::: "memory");
}

// ---------------- Flash attention: one wave per (head, 16-query tile) -------
__global__ __launch_bounds__(32)
void attn_fa(const float* __restrict__ Q, const float* __restrict__ Kg,
             const float* __restrict__ Vg, unsigned short* __restrict__ Xbf) {
  const int lane = threadIdx.x;
  const int qt = blockIdx.x, h = blockIdx.y;
  const int half = lane >> 4, ln = lane & 15;
  const int base = half ? 8 : 0;       // A-fragment K-offset base per lane half
  const int kr0  = half ? 16 : 0;      // unused: B fragments take K rows from base too

  __shared__ __align__(16) float sVf[32 * 64];   // V tile staged f32 by async DMA
  __shared__ unsigned short sP[16 * 32];
  const unsigned lbaseV = (unsigned)(uintptr_t)&sVf[0];

  // Q A-fragments for the two 32-wide K-steps over head_dim=64
  BFrag aq[2];
  {
    const float* qrow = Q + (size_t)(qt * 16 + ln) * DM + (size_t)h * HDIM;
    for (int s = 0; s < 2; ++s)
      for (int j = 0; j < 8; ++j) {
        aq[s].u[j]     = f2bf(qrow[s * 32 + base + j]);
        aq[s].u[8 + j] = f2bf(qrow[s * 32 + base + 16 + j]);
      }
  }

  v8f o0 = {}, o1 = {}, o2 = {}, o3 = {};
  float mrow[8], lrow[8];
  for (int r = 0; r < 8; ++r) { mrow[r] = -1e30f; lrow[r] = 0.0f; }

  for (int kb = 0; kb < LQ; kb += 32) {
    // kick off async V-tile copy; overlaps with scores + softmax below
    async_tile_32x64(Vg + (size_t)kb * DM + (size_t)h * HDIM, DM, lbaseV, lane);

    // scores: S = Q*K^T for 32 keys -> two 16x16 C tiles
    v8f s0 = {}, s1 = {};
    for (int t = 0; t < 2; ++t) {
      for (int s = 0; s < 2; ++s) {
        BFrag bk;
        const float* krow = Kg + (size_t)(kb + 16 * t + ln) * DM +
                            (size_t)h * HDIM + s * 32 + base;
        for (int j = 0; j < 16; ++j) bk.u[j] = f2bf(krow[j + (j & 8)]);
        if (t == 0) s0 = wmma_bf16(aq[s], bk, s0);
        else        s1 = wmma_bf16(aq[s], bk, s1);
      }
    }
    // online softmax (row stats: VGPR r covers row r (lanes 0-15) / r+8 (16-31))
    float p0[8], p1[8];
    for (int r = 0; r < 8; ++r) {
      float a = s0[r] * ATT_SCALE, b = s1[r] * ATT_SCALE;
      float v = fmaxf(a, b);
      for (int m = 8; m; m >>= 1) v = fmaxf(v, __shfl_xor(v, m, 32));
      float mn = fmaxf(mrow[r], v);
      float sc = __expf(mrow[r] - mn);
      p0[r] = __expf(a - mn);
      p1[r] = __expf(b - mn);
      float rs = p0[r] + p1[r];
      for (int m = 8; m; m >>= 1) rs += __shfl_xor(rs, m, 32);
      lrow[r] = lrow[r] * sc + rs;
      mrow[r] = mn;
      o0[r] *= sc; o1[r] *= sc; o2[r] *= sc; o3[r] *= sc;
    }
    // stage P (C-layout -> row-major LDS, bf16)
    for (int r = 0; r < 8; ++r) {
      int m = r + 8 * half;
      sP[m * 32 + ln]      = f2bf(p0[r]);
      sP[m * 32 + 16 + ln] = f2bf(p1[r]);
    }
    wait_async0();        // V tile landed in LDS
    __syncthreads();
    BFrag ap;   // P as A operand (16x32)
    for (int j = 0; j < 8; ++j) {
      ap.u[j]     = sP[ln * 32 + base + j];
      ap.u[8 + j] = sP[ln * 32 + base + 16 + j];
    }
    for (int n = 0; n < 4; ++n) {
      BFrag bv;  // V columns 16n..16n+15, K = 32 keys (f32 LDS -> bf16)
      for (int j = 0; j < 16; ++j)
        bv.u[j] = f2bf(sVf[(base + j + (j & 8)) * 64 + n * 16 + ln]);
      if      (n == 0) o0 = wmma_bf16(ap, bv, o0);
      else if (n == 1) o1 = wmma_bf16(ap, bv, o1);
      else if (n == 2) o2 = wmma_bf16(ap, bv, o2);
      else             o3 = wmma_bf16(ap, bv, o3);
    }
    __syncthreads();
  }
  for (int r = 0; r < 8; ++r) {
    int m = r + 8 * half;
    float inv = 1.0f / lrow[r];
    unsigned short* orow = Xbf + (size_t)(qt * 16 + m) * DM + (size_t)h * HDIM;
    orow[0 * 16 + ln] = f2bf(o0[r] * inv);
    orow[1 * 16 + ln] = f2bf(o1[r] * inv);
    orow[2 * 16 + ln] = f2bf(o2[r] * inv);
    orow[3 * 16 + ln] = f2bf(o3[r] * inv);
  }
}

// ---------------- Gating: top-2 of 8, softmax over selected logits ----------
__global__ __launch_bounds__(256)
void gate_k(const unsigned short* __restrict__ Xbf, const float* __restrict__ gw,
            float* __restrict__ topw, int* __restrict__ topi, int* __restrict__ counts) {
  int t = blockIdx.x * 256 + threadIdx.x;
  if (t >= LQ) return;
  float acc[NE];
  for (int e = 0; e < NE; ++e) acc[e] = 0.0f;
  const unsigned short* xr = Xbf + (size_t)t * DM;
  for (int d = 0; d < DM; ++d) {
    float x = bf2f(xr[d]);
    const float* g = gw + (size_t)d * NE;
    for (int e = 0; e < NE; ++e) acc[e] += x * g[e];
  }
  int i0 = 0; float v0 = acc[0];
  for (int e = 1; e < NE; ++e) if (acc[e] > v0) { v0 = acc[e]; i0 = e; }
  int i1 = -1; float v1 = -1e30f;
  for (int e = 0; e < NE; ++e) if (e != i0 && acc[e] > v1) { v1 = acc[e]; i1 = e; }
  float m = fmaxf(v0, v1);
  float w0 = __expf(v0 - m), w1 = __expf(v1 - m);
  float s = w0 + w1;
  topw[t * 2] = w0 / s; topw[t * 2 + 1] = w1 / s;
  topi[t * 2] = i0;     topi[t * 2 + 1] = i1;
  atomicAdd(&counts[i0], 1);
  atomicAdd(&counts[i1], 1);
}

// ---------------- 16-aligned per-expert offsets --------------------------
__global__ void offs_k(const int* __restrict__ counts, int* __restrict__ offs,
                       int* __restrict__ expert_of_slot) {
  if (threadIdx.x || blockIdx.x) return;
  int run = 0;
  for (int e = 0; e < NE; ++e) {
    offs[e] = run;
    int pad = (counts[e] + 15) & ~15;
    for (int j = 0; j < pad; ++j) expert_of_slot[run + j] = e;
    run += pad;
  }
}

// ---------------- Scatter tokens to slots --------------------------------
__global__ __launch_bounds__(256)
void scatter_k(const int* __restrict__ topi, const int* __restrict__ offs,
               int* __restrict__ cursor, int* __restrict__ token_of_slot,
               int* __restrict__ slot_of) {
  int t = blockIdx.x * 256 + threadIdx.x;
  if (t >= LQ) return;
  for (int k = 0; k < TOPK; ++k) {
    int e = topi[t * 2 + k];
    int p = atomicAdd(&cursor[e], 1);
    int s = offs[e] + p;
    token_of_slot[s] = t;
    slot_of[t * 2 + k] = s;
  }
}

// ---------------- FFN layer 1: h = gelu(x @ w1[e] + b1[e]) ---------------
// Double-buffered async weight-panel pipeline: copy panel kk+32 while the
// WMMAs consume panel kk.
__global__ __launch_bounds__(32)
void ffn1_k(const unsigned short* __restrict__ Xbf, const float* __restrict__ w1,
            const float* __restrict__ b1, const int* __restrict__ expert_of_slot,
            const int* __restrict__ token_of_slot, unsigned short* __restrict__ Hbuf) {
  const int lane = threadIdx.x, half = lane >> 4, ln = lane & 15;
  const int base = half ? 8 : 0, kr0 = half ? 16 : 0;
  const int slot0 = blockIdx.x * 16;
  const int e = expert_of_slot[slot0];      // uniform across wave
  if (e < 0) return;
  const int nbase = blockIdx.y * 64;
  const int rowtok = token_of_slot[slot0 + ln];

  __shared__ __align__(16) float sWf[2][32 * 64];   // 2 x 8KB panels
  unsigned lb[2];
  lb[0] = (unsigned)(uintptr_t)&sWf[0][0];
  lb[1] = (unsigned)(uintptr_t)&sWf[1][0];
  v8f c0 = {}, c1 = {}, c2 = {}, c3 = {};
  const size_t wb = (size_t)e * DM * HID;

  async_tile_32x64(w1 + wb + nbase, HID, lb[0], lane);   // prologue: panel kk=0
  for (int kk = 0; kk < DM; kk += 32) {
    const int cur = (kk >> 5) & 1;
    wait_async0();
    __syncthreads();
    if (kk + 32 < DM)
      async_tile_32x64(w1 + wb + (size_t)(kk + 32) * HID + nbase, HID,
                       lb[cur ^ 1], lane);
    BFrag a;
    if (rowtok >= 0) {
      const unsigned short* xr = Xbf + (size_t)rowtok * DM + kk;
      for (int j = 0; j < 8; ++j) { a.u[j] = xr[base + j]; a.u[8 + j] = xr[base + 16 + j]; }
    } else {
      for (int j = 0; j < 16; ++j) a.u[j] = 0;
    }
    const float* sw = &sWf[cur][0];
    for (int n = 0; n < 4; ++n) {
      BFrag b;
      for (int j = 0; j < 16; ++j) b.u[j] = f2bf(sw[(base + j + (j & 8)) * 64 + n * 16 + ln]);
      if      (n == 0) c0 = wmma_bf16(a, b, c0);
      else if (n == 1) c1 = wmma_bf16(a, b, c1);
      else if (n == 2) c2 = wmma_bf16(a, b, c2);
      else             c3 = wmma_bf16(a, b, c3);
    }
    __syncthreads();
  }
  for (int n = 0; n < 4; ++n) {
    int col = nbase + n * 16 + ln;
    float bias = b1[(size_t)e * HID + col];
    for (int r = 0; r < 8; ++r) {
      int m = r + 8 * half;
      float v = (n == 0 ? c0[r] : n == 1 ? c1[r] : n == 2 ? c2[r] : c3[r]) + bias;
      float g = 0.5f * v * (1.0f + erff(v * 0.70710678118f));  // exact GELU
      Hbuf[(size_t)(slot0 + m) * HID + col] = f2bf(g);
    }
  }
}

// ---------------- FFN layer 2: y = h @ w2[e] + b2[e] ---------------------
__global__ __launch_bounds__(32)
void ffn2_k(const unsigned short* __restrict__ Hbuf, const float* __restrict__ w2,
            const float* __restrict__ b2, const int* __restrict__ expert_of_slot,
            float* __restrict__ Ybuf) {
  const int lane = threadIdx.x, half = lane >> 4, ln = lane & 15;
  const int base = half ? 8 : 0, kr0 = half ? 16 : 0;
  const int slot0 = blockIdx.x * 16;
  const int e = expert_of_slot[slot0];
  if (e < 0) return;
  const int nbase = blockIdx.y * 64;

  __shared__ __align__(16) float sWf[2][32 * 64];
  unsigned lb[2];
  lb[0] = (unsigned)(uintptr_t)&sWf[0][0];
  lb[1] = (unsigned)(uintptr_t)&sWf[1][0];
  v8f c0 = {}, c1 = {}, c2 = {}, c3 = {};
  const size_t wb = (size_t)e * HID * DM;
  const unsigned short* hr0 = Hbuf + (size_t)(slot0 + ln) * HID;

  async_tile_32x64(w2 + wb + nbase, DM, lb[0], lane);
  for (int kk = 0; kk < HID; kk += 32) {
    const int cur = (kk >> 5) & 1;
    wait_async0();
    __syncthreads();
    if (kk + 32 < HID)
      async_tile_32x64(w2 + wb + (size_t)(kk + 32) * DM + nbase, DM,
                       lb[cur ^ 1], lane);
    BFrag a;
    for (int j = 0; j < 8; ++j) {
      a.u[j]     = hr0[kk + base + j];
      a.u[8 + j] = hr0[kk + base + 16 + j];
    }
    const float* sw = &sWf[cur][0];
    for (int n = 0; n < 4; ++n) {
      BFrag b;
      for (int j = 0; j < 16; ++j) b.u[j] = f2bf(sw[(base + j + (j & 8)) * 64 + n * 16 + ln]);
      if      (n == 0) c0 = wmma_bf16(a, b, c0);
      else if (n == 1) c1 = wmma_bf16(a, b, c1);
      else if (n == 2) c2 = wmma_bf16(a, b, c2);
      else             c3 = wmma_bf16(a, b, c3);
    }
    __syncthreads();
  }
  for (int n = 0; n < 4; ++n) {
    int col = nbase + n * 16 + ln;
    float bias = b2[(size_t)e * DM + col];
    for (int r = 0; r < 8; ++r) {
      int m = r + 8 * half;
      float v = (n == 0 ? c0[r] : n == 1 ? c1[r] : n == 2 ? c2[r] : c3[r]) + bias;
      Ybuf[(size_t)(slot0 + m) * DM + col] = v;
    }
  }
}

// ---------------- Weighted combine ---------------------------------------
__global__ __launch_bounds__(256)
void comb_k(const float* __restrict__ Ybuf, const float* __restrict__ topw,
            const int* __restrict__ slot_of, float* __restrict__ out) {
  size_t i = (size_t)blockIdx.x * 256 + threadIdx.x;
  if (i >= (size_t)LQ * DM) return;
  int t = (int)(i / DM), d = (int)(i % DM);
  float r = topw[t * 2]     * Ybuf[(size_t)slot_of[t * 2]     * DM + d] +
            topw[t * 2 + 1] * Ybuf[(size_t)slot_of[t * 2 + 1] * DM + d];
  out[i] = r;
}

extern "C" void kernel_launch(void* const* d_in, const int* in_sizes, int n_in,
                              void* d_out, int out_size, void* d_ws, size_t ws_size,
                              hipStream_t stream) {
  (void)in_sizes; (void)n_in; (void)out_size; (void)ws_size;
  const float* Q  = (const float*)d_in[0];
  const float* K  = (const float*)d_in[1];
  const float* V  = (const float*)d_in[2];
  const float* gw = (const float*)d_in[3];
  const float* w1 = (const float*)d_in[4];
  const float* b1 = (const float*)d_in[5];
  const float* w2 = (const float*)d_in[6];
  const float* b2 = (const float*)d_in[7];
  float* out = (float*)d_out;

  uint8_t* p = (uint8_t*)d_ws;
  unsigned short* Xbf  = (unsigned short*)p; p += (size_t)LQ * DM * 2;      // 8 MB
  unsigned short* Hbuf = (unsigned short*)p; p += (size_t)NSLOT * HID * 2;  // ~69 MB
  float* Ybuf = (float*)p;                   p += (size_t)NSLOT * DM * 4;   // ~35 MB
  float* topw = (float*)p;                   p += (size_t)LQ * 2 * 4;
  int* topi = (int*)p;                       p += (size_t)LQ * 2 * 4;
  int* slot_of = (int*)p;                    p += (size_t)LQ * 2 * 4;
  int* token_of_slot = (int*)p;              p += (size_t)NSLOT * 4;
  int* expert_of_slot = (int*)p;             p += (size_t)NSLOT * 4;
  int* counts = (int*)p;                     p += NE * 4;
  int* offs = (int*)p;                       p += NE * 4;
  int* cursor = (int*)p;                     p += NE * 4;

  hipMemsetAsync(counts, 0, NE * 4, stream);
  hipMemsetAsync(cursor, 0, NE * 4, stream);
  hipMemsetAsync(token_of_slot, 0xFF, NSLOT * 4, stream);
  hipMemsetAsync(expert_of_slot, 0xFF, NSLOT * 4, stream);

  attn_fa<<<dim3(LQ / 16, NH), 32, 0, stream>>>(Q, K, V, Xbf);
  gate_k<<<LQ / 256, 256, 0, stream>>>(Xbf, gw, topw, topi, counts);
  offs_k<<<1, 1, 0, stream>>>(counts, offs, expert_of_slot);
  scatter_k<<<LQ / 256, 256, 0, stream>>>(topi, offs, cursor, token_of_slot, slot_of);
  ffn1_k<<<dim3(NTILES, HID / 64), 32, 0, stream>>>(Xbf, w1, b1, expert_of_slot,
                                                    token_of_slot, Hbuf);
  ffn2_k<<<dim3(NTILES, DM / 64), 32, 0, stream>>>(Hbuf, w2, b2, expert_of_slot, Ybuf);
  comb_k<<<(LQ * DM) / 256, 256, 0, stream>>>(Ybuf, topw, slot_of, out);
}
